// KAttDec_20203526160857
// MI455X (gfx1250) — matrix operand order repaired and checked by reading, not compiled, not run on hardware
//
#include <hip/hip_runtime.h>
#include <hip/hip_bf16.h>

// ---------------------------------------------------------------------------
// Problem constants (match reference: B=4, N=2048, E=512, K=16, H=8, EMB=1024)
// ---------------------------------------------------------------------------
constexpr int Bc   = 4;
constexpr int Nc   = 2048;
constexpr int Ec   = 512;
constexpr int Kc   = 16;
constexpr int Hc   = 8;
constexpr int EMBc = 1024;
constexpr int Dc   = 128;   // head dim

typedef __attribute__((ext_vector_type(16))) _Float16 half16;
typedef __attribute__((ext_vector_type(8)))  _Float16 half8_t;
typedef __attribute__((ext_vector_type(8)))  float    float8;

union FragU { half16 v; half8_t h[2]; };

// ---------------------------------------------------------------------------
// gfx1250 async global->LDS copy (ASYNCcnt path), guarded so the file compiles
// on toolchains without the builtins (falls back to the sync VGPR round-trip).
// Probe-discovered signature: (int4 addrspace(1)*, int4 addrspace(3)*, Ii, Ii)
// ---------------------------------------------------------------------------
#if defined(__AMDGCN__) &&                                              \
    __has_builtin(__builtin_amdgcn_global_load_async_to_lds_b128) &&    \
    __has_builtin(__builtin_amdgcn_s_wait_asynccnt)
#define HAS_ASYNC_LDS 1
#else
#define HAS_ASYNC_LDS 0
#endif

#if HAS_ASYNC_LDS
typedef __attribute__((ext_vector_type(4))) int int4v;
typedef __attribute__((address_space(1))) int4v gint4_t;  // global (__device__)
typedef __attribute__((address_space(3))) int4v lint4_t;  // LDS (__shared__)
__device__ __forceinline__ void async_ld_b128(const void* g, void* l) {
  __builtin_amdgcn_global_load_async_to_lds_b128((gint4_t*)g, (lint4_t*)l, 0, 0);
}
__device__ __forceinline__ void async_wait0() {
  __builtin_amdgcn_s_wait_asynccnt(0);
}
#endif

__device__ inline float8 f8zero() {
  float8 z;
#pragma unroll
  for (int i = 0; i < 8; ++i) z[i] = 0.0f;
  return z;
}

__device__ inline float8 wmma_f16(half16 a, half16 b, float8 c) {
  // v_wmma_f32_16x16x32_f16: D = A(16x32) * B(32x16) + C(16x16 f32)
  return __builtin_amdgcn_wmma_f32_16x16x32_f16(false, a, false, b, (short)0, c,
                                                false, false);
}

// A fragment: 16(M) x 32(K) f16 from row-major buffer (row stride `ld` halves).
// Per ISA 7.12.2: lane = M (mod 16); half-wave selects K-octet; VGPRs 0-3 hold
// K = hi*8..+7, VGPRs 4-7 hold K = 16+hi*8..+7  -> two 16B loads per lane.
__device__ inline half16 load_frag_a(const _Float16* p, int ld) {
  const int lane = threadIdx.x & 31;
  const int row  = lane & 15;
  const int hi   = (lane >> 4) & 1;
  const _Float16* r = p + (long)row * ld + hi * 8;
  FragU f;
  f.h[0] = *(const half8_t*)(r);
  f.h[1] = *(const half8_t*)(r + 16);
  return f.v;
}

// B fragment: 32(K) x 16(N) f16, loaded from B^T stored row-major (N x K).
// Lane = N (mod 16); half-wave selects K-octet: elements 0-7 hold
// K = hi*8..+7, elements 8-15 hold K = 16+hi*8..+7 (same as A fragment).
__device__ inline half16 load_frag_b(const _Float16* p, int ld) {
  const int lane = threadIdx.x & 31;
  const int row  = lane & 15;
  const int hi   = (lane >> 4) & 1;
  const _Float16* r = p + (long)row * ld + hi * 8;
  FragU f;
  f.h[0] = *(const half8_t*)(r);
  f.h[1] = *(const half8_t*)(r + 16);
  return f.v;
}

// ---------------------------------------------------------------------------
// Small data-movement kernels
// ---------------------------------------------------------------------------
__global__ void k_f32toh(const float* __restrict__ s, _Float16* __restrict__ d,
                         long n) {
  long i = (long)blockIdx.x * blockDim.x + threadIdx.x;
  long stride = (long)gridDim.x * blockDim.x;
  for (; i < n; i += stride) d[i] = (_Float16)s[i];
}

// x (B,N,E) f32 -> xt (B,E,N) in both f16 (for WMMA Gram) and f32 (exact means)
__global__ void k_xpose(const float* __restrict__ x, _Float16* __restrict__ xt_h,
                        float* __restrict__ xt_f) {
  __shared__ float tile[32][33];
  const int e0 = blockIdx.x * 32, n0 = blockIdx.y * 32, b = blockIdx.z;
#pragma unroll
  for (int i = 0; i < 4; ++i) {
    int n = n0 + threadIdx.y + i * 8;
    tile[threadIdx.y + i * 8][threadIdx.x] =
        x[((long)b * Nc + n) * Ec + e0 + threadIdx.x];
  }
  __syncthreads();
#pragma unroll
  for (int i = 0; i < 4; ++i) {
    int e = e0 + threadIdx.y + i * 8;
    long o = ((long)b * Ec + e) * Nc + n0 + threadIdx.x;
    float v = tile[threadIdx.x][threadIdx.y + i * 8];
    xt_f[o] = v;
    xt_h[o] = (_Float16)v;
  }
}

// out[b,n,2e] = x[b,n,e]; q_h likewise (even interleave of q)
__global__ void k_copyx(const float* __restrict__ x, float* __restrict__ out,
                        _Float16* __restrict__ q_h) {
  long i = (long)blockIdx.x * 256 + threadIdx.x;  // total = B*N*E (mult of 256)
  int e = (int)(i % Ec);
  long t = i / Ec;
  int n = (int)(t % Nc);
  int b = (int)(t / Nc);
  float v = x[i];
  out[(long)b * (2 * Nc) * EMBc + (long)n * EMBc + 2 * e] = v;
  q_h[((long)b * Nc + n) * EMBc + 2 * e] = (_Float16)v;
}

// ---------------------------------------------------------------------------
// Generic WMMA GEMM:  C[m,o] = sum_k A[m,k] * Bsrc[o,k]   (A @ Bsrc^T)
// Block tile 128x128, BK=32, 8 waves each computing 32(M) x 64(N).
// Double-buffered async global->LDS staging when available.
// MODE 0: f32 store to gram (per-z batch); MODE 1: +bias, f16 scatter into
// head-major QKV (s,B,H,N,128) with column offset obase; MODE 2: +bias, f32
// store into output rows N..2N.
// ---------------------------------------------------------------------------
template <int MODE>
__global__ __launch_bounds__(256) void k_gemm(
    const _Float16* __restrict__ A, const _Float16* __restrict__ Bsrc,
    int lda, int ldb, int kdim, long sA, long sB,
    const float* __restrict__ bias, float* __restrict__ outF,
    _Float16* __restrict__ outH, int obase) {
#if HAS_ASYNC_LDS
  __shared__ _Float16 As[2][128 * 40];  // 80B rows: 16B-aligned, 16 consecutive
  __shared__ _Float16 Bs[2][128 * 40];  // rows hit distinct bank groups
#else
  __shared__ _Float16 As[1][128 * 40];
  __shared__ _Float16 Bs[1][128 * 40];
#endif

  const int tid = threadIdx.x;
  const int lane = tid & 31, w = tid >> 5;
  const int ln = lane & 15, hi = (lane >> 4) & 1;
  const int wm = w & 3, wn = w >> 2;  // 4 waves along M, 2 along N
  const int m0 = blockIdx.x * 128, n0 = blockIdx.y * 128;
  const int z = blockIdx.z;
  A += (long)z * sA;
  Bsrc += (long)z * sB;

  float8 acc[2][4];
#pragma unroll
  for (int i = 0; i < 2; ++i)
#pragma unroll
    for (int j = 0; j < 4; ++j) acc[i][j] = f8zero();

#if HAS_ASYNC_LDS
  // ---- double-buffered async pipeline ----
#pragma unroll
  for (int i = 0; i < 2; ++i) {
    int c = tid + i * 256;  // 512 chunks of 8 halves per matrix
    int row = c >> 2, c8 = (c & 3) * 8;
    async_ld_b128(&A[(long)(m0 + row) * lda + c8], &As[0][row * 40 + c8]);
    async_ld_b128(&Bsrc[(long)(n0 + row) * ldb + c8], &Bs[0][row * 40 + c8]);
  }
  async_wait0();
  __syncthreads();
  int cur = 0;
  for (int k0 = 0; k0 < kdim; k0 += 32) {
    const int nxt = cur ^ 1;
    if (k0 + 32 < kdim) {  // prefetch next K tile while WMMAs run
#pragma unroll
      for (int i = 0; i < 2; ++i) {
        int c = tid + i * 256;
        int row = c >> 2, c8 = (c & 3) * 8;
        async_ld_b128(&A[(long)(m0 + row) * lda + k0 + 32 + c8],
                      &As[nxt][row * 40 + c8]);
        async_ld_b128(&Bsrc[(long)(n0 + row) * ldb + k0 + 32 + c8],
                      &Bs[nxt][row * 40 + c8]);
      }
    }
    half16 af[2], bf[4];
#pragma unroll
    for (int i = 0; i < 2; ++i)
      af[i] = load_frag_a(&As[cur][(wm * 32 + i * 16) * 40], 40);
#pragma unroll
    for (int j = 0; j < 4; ++j)
      bf[j] = load_frag_b(&Bs[cur][(wn * 64 + j * 16) * 40], 40);
#pragma unroll
    for (int i = 0; i < 2; ++i)
#pragma unroll
      for (int j = 0; j < 4; ++j) acc[i][j] = wmma_f16(af[i], bf[j], acc[i][j]);
    async_wait0();
    __syncthreads();
    cur = nxt;
  }
#else
  // ---- fallback: synchronous staging ----
  for (int k0 = 0; k0 < kdim; k0 += 32) {
    __syncthreads();
#pragma unroll
    for (int i = 0; i < 2; ++i) {
      int c = tid + i * 256;
      int row = c >> 2, c8 = (c & 3) * 8;
      *(half8_t*)&As[0][row * 40 + c8] =
          *(const half8_t*)&A[(long)(m0 + row) * lda + k0 + c8];
      *(half8_t*)&Bs[0][row * 40 + c8] =
          *(const half8_t*)&Bsrc[(long)(n0 + row) * ldb + k0 + c8];
    }
    if (k0 + 32 < kdim) {  // gfx1250 global_prefetch of next K tile
      int row = tid >> 1;
      __builtin_prefetch(&A[(long)(m0 + row) * lda + k0 + 32], 0, 1);
      __builtin_prefetch(&Bsrc[(long)(n0 + row) * ldb + k0 + 32], 0, 1);
    }
    __syncthreads();

    half16 af[2], bf[4];
#pragma unroll
    for (int i = 0; i < 2; ++i)
      af[i] = load_frag_a(&As[0][(wm * 32 + i * 16) * 40], 40);
#pragma unroll
    for (int j = 0; j < 4; ++j)
      bf[j] = load_frag_b(&Bs[0][(wn * 64 + j * 16) * 40], 40);
#pragma unroll
    for (int i = 0; i < 2; ++i)
#pragma unroll
      for (int j = 0; j < 4; ++j) acc[i][j] = wmma_f16(af[i], bf[j], acc[i][j]);
  }
#endif

  // Epilogue. C layout: lane -> col (ln), half-wave+VGPR -> row (hi*8+r).
#pragma unroll
  for (int i = 0; i < 2; ++i) {
#pragma unroll
    for (int j = 0; j < 4; ++j) {
      const int tm = m0 + wm * 32 + i * 16 + hi * 8;
      const int tn = n0 + wn * 64 + j * 16 + ln;
#pragma unroll
      for (int r = 0; r < 8; ++r) {
        float v = acc[i][j][r];
        int gm = tm + r, gn = tn;
        if (MODE == 0) {
          outF[(long)z * Ec * Ec + (long)gm * Ec + gn] = v;
        } else if (MODE == 1) {
          int o = obase + gn;
          float vb = v + bias[o];
          int s = o >> 10, hh = (o >> 7) & (Hc - 1), dd = o & (Dc - 1);
          int bb = gm >> 11, n = gm & (Nc - 1);
          outH[((((long)s * Bc + bb) * Hc + hh) * Nc + n) * Dc + dd] =
              (_Float16)vb;
        } else {
          float vb = v + bias[gn];
          int bb = gm >> 11, n = gm & (Nc - 1);
          outF[(long)bb * (2 * Nc) * EMBc + (long)(Nc + n) * EMBc + gn] = vb;
        }
      }
    }
  }
}

// ---------------------------------------------------------------------------
// kNN: top-16 smallest distances per (b, e) channel row from the Gram matrix
// dist(e,f) = g[e,e] + g[f,f] - 2 g[e,f]
// ---------------------------------------------------------------------------
__global__ __launch_bounds__(256) void k_topk(const float* __restrict__ gram,
                                              int* __restrict__ idxs) {
  __shared__ float diag[Ec];
  const int t = blockIdx.x * 256 + threadIdx.x;  // one (b,e) per thread
  const int b = t / Ec, e = t % Ec;              // blocks never straddle b
  const float* g = gram + (long)b * Ec * Ec;
  for (int f = threadIdx.x; f < Ec; f += 256) diag[f] = g[(long)f * Ec + f];
  __syncthreads();

  float bd[Kc];
  int bi[Kc];
#pragma unroll
  for (int i = 0; i < Kc; ++i) { bd[i] = 3.4e38f; bi[i] = 0; }
  const float ge = diag[e];
  for (int f = 0; f < Ec; ++f) {
    float d = ge + diag[f] - 2.0f * g[(long)e * Ec + f];
    if (d < bd[Kc - 1]) {
      int j = Kc - 1;
      while (j > 0 && bd[j - 1] > d) { bd[j] = bd[j - 1]; bi[j] = bi[j - 1]; --j; }
      bd[j] = d;
      bi[j] = f;
    }
  }
#pragma unroll
  for (int i = 0; i < Kc; ++i) idxs[(long)t * Kc + i] = bi[i];
}

// mean over the 16 neighbor channel rows; writes odd interleave of q
__global__ __launch_bounds__(256) void k_knnmean(const float* __restrict__ xt_f,
                                                 const int* __restrict__ idxs,
                                                 float* __restrict__ out,
                                                 _Float16* __restrict__ q_h) {
  const int be = blockIdx.x;
  const int b = be / Ec, e = be % Ec;
  __shared__ int id[Kc];
  if (threadIdx.x < Kc) id[threadIdx.x] = idxs[(long)be * Kc + threadIdx.x];
  __syncthreads();
  const float* xb = xt_f + (long)b * Ec * Nc;
  for (int n = threadIdx.x; n < Nc; n += 256) {
    float s = 0.0f;
#pragma unroll
    for (int j = 0; j < Kc; ++j) s += xb[(long)id[j] * Nc + n];
    float mean = s * (1.0f / (float)Kc);
    out[(long)b * (2 * Nc) * EMBc + (long)n * EMBc + 2 * e + 1] = mean;
    q_h[((long)b * Nc + n) * EMBc + 2 * e + 1] = (_Float16)mean;
  }
}

// ---------------------------------------------------------------------------
// Flash attention: one block per (b, h, 128 query rows). 8 waves, each owns a
// 16-row query strip. S = Q K^T (WMMA), online softmax fully in registers
// (S and O share the same C-layout row->lane mapping, so the running-max
// rescale is lane-local; row reductions are 4 half-wave shfl_xor steps),
// P written to LDS as the A-operand of P @ V (WMMA). Q/K tiles staged via
// async global->LDS when available (wait overlaps the V transpose VALU work).
// ---------------------------------------------------------------------------
constexpr int AT_LD = Dc + 8;  // 136 halves = 272B rows -> conflict-free b128

__global__ __launch_bounds__(256) void k_attn(const _Float16* __restrict__ qkv,
                                              _Float16* __restrict__ o_h) {
  extern __shared__ _Float16 smem[];
  _Float16* Qs = smem;                  // 128 x 136
  _Float16* Ks = Qs + 128 * AT_LD;      // 128 x 136
  _Float16* VTs = Ks + 128 * AT_LD;     // V transposed: [d][key]
  _Float16* Ps = VTs + 128 * AT_LD;     // probs f16: [qrow][key]

  const int tid = threadIdx.x, lane = tid & 31, w = tid >> 5;
  const int ln = lane & 15, hi = (lane >> 4) & 1;
  const int qt = blockIdx.x, h = blockIdx.y, b = blockIdx.z;

  const long plane = (long)Nc * Dc;
  const _Float16* Qg = qkv + (((long)0 * Bc + b) * Hc + h) * plane + (long)qt * 128 * Dc;
  const _Float16* Kg = qkv + (((long)1 * Bc + b) * Hc + h) * plane;
  const _Float16* Vg = qkv + (((long)2 * Bc + b) * Hc + h) * plane;

  // stage Q tile once
#pragma unroll
  for (int i = 0; i < 8; ++i) {
    int c = tid + i * 256;                // 2048 chunks (128 rows x 16 octets)
    int row = c >> 4, c8 = (c & 15) * 8;
#if HAS_ASYNC_LDS
    async_ld_b128(&Qg[(long)row * Dc + c8], &Qs[row * AT_LD + c8]);
#else
    *(half8_t*)&Qs[row * AT_LD + c8] = *(const half8_t*)&Qg[(long)row * Dc + c8];
#endif
  }

  float m_i[8], l_i[8];
  float8 accO[8];
#pragma unroll
  for (int r = 0; r < 8; ++r) { m_i[r] = -1e30f; l_i[r] = 0.0f; }
#pragma unroll
  for (int t = 0; t < 8; ++t) accO[t] = f8zero();

  const float scale = 0.08838834764831845f;  // 1/sqrt(128)

  for (int kb = 0; kb < Nc / 128; ++kb) {
    __syncthreads();  // prior reads of Ks/VTs complete before restage
#pragma unroll
    for (int i = 0; i < 8; ++i) {
      int c = tid + i * 256;
      int row = c >> 4, c8 = (c & 15) * 8;
#if HAS_ASYNC_LDS
      async_ld_b128(&Kg[(long)(kb * 128 + row) * Dc + c8],
                    &Ks[row * AT_LD + c8]);
#else
      *(half8_t*)&Ks[row * AT_LD + c8] =
          *(const half8_t*)&Kg[(long)(kb * 128 + row) * Dc + c8];
#endif
      half8_t vv = *(const half8_t*)&Vg[(long)(kb * 128 + row) * Dc + c8];
#pragma unroll
      for (int j = 0; j < 8; ++j) VTs[(c8 + j) * AT_LD + row] = vv[j];
    }
#if HAS_ASYNC_LDS
    async_wait0();  // also covers the Q-tile asyncs on the first iteration
#endif
    __syncthreads();

    // S strip = Q[w*16..+16) @ K^T  (8 col tiles x 4 K-steps)
    half16 aq[4];
#pragma unroll
    for (int kk = 0; kk < 4; ++kk)
      aq[kk] = load_frag_a(&Qs[(w * 16) * AT_LD + kk * 32], AT_LD);
    float8 accS[8];
#pragma unroll
    for (int t = 0; t < 8; ++t) {
      accS[t] = f8zero();
#pragma unroll
      for (int kk = 0; kk < 4; ++kk) {
        half16 bk = load_frag_b(&Ks[(t * 16) * AT_LD + kk * 32], AT_LD);
        accS[t] = wmma_f16(aq[kk], bk, accS[t]);
      }
    }

    // online softmax; row m = hi*8 + r lives on the 16 lanes of this half-wave
    float mnew[8], corr[8];
#pragma unroll
    for (int r = 0; r < 8; ++r) {
      float mv = -1e30f;
#pragma unroll
      for (int t = 0; t < 8; ++t) mv = fmaxf(mv, accS[t][r]);
      mv = fmaxf(mv, __shfl_xor(mv, 1, 32));
      mv = fmaxf(mv, __shfl_xor(mv, 2, 32));
      mv = fmaxf(mv, __shfl_xor(mv, 4, 32));
      mv = fmaxf(mv, __shfl_xor(mv, 8, 32));
      mv *= scale;
      mnew[r] = fmaxf(m_i[r], mv);
      corr[r] = __expf(m_i[r] - mnew[r]);
      m_i[r] = mnew[r];
    }
#pragma unroll
    for (int t = 0; t < 8; ++t)
#pragma unroll
      for (int r = 0; r < 8; ++r)
        accS[t][r] = __expf(accS[t][r] * scale - mnew[r]);
#pragma unroll
    for (int r = 0; r < 8; ++r) {
      float sv = 0.0f;
#pragma unroll
      for (int t = 0; t < 8; ++t) sv += accS[t][r];
      sv += __shfl_xor(sv, 1, 32);
      sv += __shfl_xor(sv, 2, 32);
      sv += __shfl_xor(sv, 4, 32);
      sv += __shfl_xor(sv, 8, 32);
      l_i[r] = l_i[r] * corr[r] + sv;
    }
#pragma unroll
    for (int t = 0; t < 8; ++t)
#pragma unroll
      for (int r = 0; r < 8; ++r) accO[t][r] *= corr[r];

    // stash probs (each wave writes/reads only its own 16 rows of Ps)
#pragma unroll
    for (int t = 0; t < 8; ++t)
#pragma unroll
      for (int r = 0; r < 8; ++r)
        Ps[(w * 16 + hi * 8 + r) * AT_LD + t * 16 + ln] = (_Float16)accS[t][r];

    // O += P @ V
    half16 ap[4];
#pragma unroll
    for (int kk = 0; kk < 4; ++kk)
      ap[kk] = load_frag_a(&Ps[(w * 16) * AT_LD + kk * 32], AT_LD);
#pragma unroll
    for (int t = 0; t < 8; ++t)
#pragma unroll
      for (int kk = 0; kk < 4; ++kk) {
        half16 bv = load_frag_b(&VTs[(t * 16) * AT_LD + kk * 32], AT_LD);
        accO[t] = wmma_f16(ap[kk], bv, accO[t]);
      }
  }

  // normalize and write head output (B, N, EMB) f16
#pragma unroll
  for (int t = 0; t < 8; ++t) {
#pragma unroll
    for (int r = 0; r < 8; ++r) {
      int row = qt * 128 + w * 16 + hi * 8 + r;
      int col = t * 16 + ln;
      float v = accO[t][r] / l_i[r];
      o_h[((long)b * Nc + row) * EMBc + h * Dc + col] = (_Float16)v;
    }
  }
}

// ---------------------------------------------------------------------------
// Host-side orchestration
// ---------------------------------------------------------------------------
extern "C" void kernel_launch(void* const* d_in, const int* in_sizes, int n_in,
                              void* d_out, int out_size, void* d_ws,
                              size_t ws_size, hipStream_t stream) {
  (void)in_sizes; (void)n_in; (void)out_size; (void)ws_size;
  const float* x     = (const float*)d_in[0];
  const float* x_enc = (const float*)d_in[1];
  const float* w_in  = (const float*)d_in[2];
  const float* b_in  = (const float*)d_in[3];
  const float* w_out = (const float*)d_in[4];
  const float* b_out = (const float*)d_in[5];
  float* out = (float*)d_out;

  char* p = (char*)d_ws;
  auto take = [&](size_t bytes) {
    char* q = p;
    p += (bytes + 255) & ~(size_t)255;
    return q;
  };
  _Float16* xt_h    = (_Float16*)take((size_t)Bc * Ec * Nc * 2);
  float*    xt_f    = (float*)take((size_t)Bc * Ec * Nc * 4);
  float*    gram    = (float*)take((size_t)Bc * Ec * Ec * 4);
  _Float16* xenc_h  = (_Float16*)take((size_t)Bc * Nc * EMBc * 2);
  _Float16* w_in_h  = (_Float16*)take((size_t)3 * EMBc * EMBc * 2);
  _Float16* w_out_h = (_Float16*)take((size_t)EMBc * EMBc * 2);
  _Float16* q_h     = (_Float16*)take((size_t)Bc * Nc * EMBc * 2);
  _Float16* qkv_h   = (_Float16*)take((size_t)3 * Bc * Nc * EMBc * 2);
  _Float16* o_h     = (_Float16*)take((size_t)Bc * Nc * EMBc * 2);
  int*      idxs    = (int*)take((size_t)Bc * Ec * Kc * 4);

  // 1) precision converts + channel transpose
  k_f32toh<<<4096, 256, 0, stream>>>(x_enc, xenc_h, (long)Bc * Nc * EMBc);
  k_f32toh<<<2048, 256, 0, stream>>>(w_in, w_in_h, (long)3 * EMBc * EMBc);
  k_f32toh<<<1024, 256, 0, stream>>>(w_out, w_out_h, (long)EMBc * EMBc);
  k_xpose<<<dim3(Ec / 32, Nc / 32, Bc), dim3(32, 8), 0, stream>>>(x, xt_h, xt_f);

  // 2) Gram matrix (per batch): xt @ xt^T, K = 2048
  k_gemm<0><<<dim3(Ec / 128, Ec / 128, Bc), 256, 0, stream>>>(
      xt_h, xt_h, Nc, Nc, Nc, (long)Ec * Nc, (long)Ec * Nc, nullptr, gram,
      nullptr, 0);

  // 3) kNN top-16 + neighbor means; build interleaved q (also output rows 0..N)
  k_topk<<<(Bc * Ec) / 256, 256, 0, stream>>>(gram, idxs);
  k_knnmean<<<Bc * Ec, 256, 0, stream>>>(xt_f, idxs, out, q_h);
  k_copyx<<<(int)(((long)Bc * Nc * Ec) / 256), 256, 0, stream>>>(x, out, q_h);

  // 4) projections: Q = q @ Wq^T + bq ; K,V = x_enc @ [Wk;Wv]^T + b
  k_gemm<1><<<dim3((Bc * Nc) / 128, EMBc / 128, 1), 256, 0, stream>>>(
      q_h, w_in_h, EMBc, EMBc, EMBc, 0, 0, b_in, nullptr, qkv_h, 0);
  k_gemm<1><<<dim3((Bc * Nc) / 128, (2 * EMBc) / 128, 1), 256, 0, stream>>>(
      xenc_h, w_in_h + (size_t)EMBc * EMBc, EMBc, EMBc, EMBc, 0, 0, b_in,
      nullptr, qkv_h, EMBc);

  // 5) flash attention per (b, h); dynamic LDS = 4 tiles of 128x136 f16
  size_t attn_lds = (size_t)4 * 128 * AT_LD * sizeof(_Float16);  // 139264 B
  k_attn<<<dim3(Nc / 128, Hc, Bc), 256, attn_lds, stream>>>(qkv_h, o_h);

  // 6) output projection -> rows N..2N of output (f32)
  k_gemm<2><<<dim3((Bc * Nc) / 128, EMBc / 128, 1), 256, 0, stream>>>(
      o_h, w_out_h, EMBc, EMBc, EMBc, 0, 0, b_out, out, nullptr, 0);
}
